// SelfAttention_2817498546673
// MI455X (gfx1250) — matrix operand order repaired and checked by reading, not compiled, not run on hardware
//
#include <hip/hip_runtime.h>
#include <hip/hip_bf16.h>

// ---------------------------------------------------------------------------
// Self-attention for MI455X (gfx1250): all matmuls via v_wmma_f32_16x16x32_bf16
// LDS-tiled flash attention (async-to-LDS triple buffering) + 4x4-tile GEMM.
// ---------------------------------------------------------------------------

typedef __attribute__((ext_vector_type(16))) __bf16 v16bf;
typedef __attribute__((ext_vector_type(8)))  __bf16 v8bf;
typedef __attribute__((ext_vector_type(8)))  float  v8f;

#define D_MODEL 1024
#define SEQ     2048
#define NHEAD   16
#define HDIM    64
#define BATCH   4
#define MROWS   (BATCH * SEQ)   // 8192

static __device__ __forceinline__ v16bf afrag(const __bf16* p) {
    // A-matrix 16x32 bf16 fragment: lane supplies M=lane%16; caller passes
    // p = rowptr + k0 + 8*half. K(e) = 8*half + e for e<8, +16 more for e>=8.
    v8bf lo = *(const v8bf*)(p);
    v8bf hi = *(const v8bf*)(p + 16);
    return __builtin_shufflevector(lo, hi, 0,1,2,3,4,5,6,7,8,9,10,11,12,13,14,15);
}

static __device__ __forceinline__ v8f wmma_bf16(v16bf a, v16bf b, v8f c) {
    return __builtin_amdgcn_wmma_f32_16x16x32_bf16(false, a, false, b,
                                                   (short)0, c, false, false);
}

// --------------------------- conversion kernels ----------------------------

__global__ __launch_bounds__(256) void f32_to_bf16_k(const float* __restrict__ in,
                                                     __bf16* __restrict__ out, int n) {
    int i = blockIdx.x * 256 + threadIdx.x;
    if (i < n) out[i] = (__bf16)in[i];
}

// Wt[n][k] = W[k][n]  (both 1024x1024), output bf16
__global__ __launch_bounds__(256) void transpose_bf16_k(const float* __restrict__ W,
                                                        __bf16* __restrict__ Wt) {
    int i = blockIdx.x * 256 + threadIdx.x;            // linear index into Wt
    int nrow = i >> 10;
    int k    = i & 1023;
    Wt[i] = (__bf16)W[k * 1024 + nrow];
}

// ------------------------------ projection GEMM ----------------------------
// C[M=8192, N=1024] = A_bf16[M,K=1024] @ Wt_bf16[N,K]^T + bias
// Each wave owns a 64x64 output block (4x4 WMMA tiles): per k-step,
// 4 A-frags x 4 B-frags -> 16 WMMAs, amortizing both operand streams.
// MODE 0: store bf16 row-major   (Q, K)
// MODE 1: store bf16 transposed V layout Vt[(b*16+h)*64+d][s]
// MODE 2: store f32 row-major    (final output)
template <int MODE>
__global__ __launch_bounds__(256) void gemm_proj(const __bf16* __restrict__ A,
                                                 const __bf16* __restrict__ Bt,
                                                 const float*  __restrict__ bias,
                                                 void* __restrict__ outp) {
    const int wid  = (blockIdx.x * 256 + threadIdx.x) >> 5;  // 0..2047
    const int lane = threadIdx.x & 31;
    const int ln   = lane & 15;
    const int hf   = lane >> 4;
    const int mg   = wid >> 4;       // 0..127 : 64-row group
    const int ng   = wid & 15;       // 0..15  : 64-col group

    size_t arow[4];
#pragma unroll
    for (int mi = 0; mi < 4; ++mi)
        arow[mi] = (size_t)(mg * 64 + mi * 16 + ln) * D_MODEL;

    int    ncol[4];
    size_t brow[4];
    v8f    acc[4][4];
#pragma unroll
    for (int j = 0; j < 4; ++j) {
        ncol[j] = ng * 64 + j * 16 + ln;
        brow[j] = (size_t)ncol[j] * D_MODEL;
        float bz = bias[ncol[j]];
#pragma unroll
        for (int mi = 0; mi < 4; ++mi)
#pragma unroll
            for (int r = 0; r < 8; ++r) acc[mi][j][r] = bz;
    }

#pragma unroll 1
    for (int k0 = 0; k0 < D_MODEL; k0 += 32) {
        v16bf a[4];
#pragma unroll
        for (int mi = 0; mi < 4; ++mi) a[mi] = afrag(A + arow[mi] + k0 + 8 * hf);
#pragma unroll
        for (int j = 0; j < 4; ++j) {
            v16bf b = afrag(Bt + brow[j] + k0 + 8 * hf);
#pragma unroll
            for (int mi = 0; mi < 4; ++mi)
                acc[mi][j] = wmma_bf16(a[mi], b, acc[mi][j]);
        }
    }

#pragma unroll
    for (int mi = 0; mi < 4; ++mi)
#pragma unroll
        for (int j = 0; j < 4; ++j)
#pragma unroll
            for (int r = 0; r < 8; ++r) {
                const int orow = mg * 64 + mi * 16 + r + 8 * hf;  // C row = r+8*half
                const int ocol = ncol[j];
                const float v  = acc[mi][j][r];
                if (MODE == 2) {
                    ((float*)outp)[(size_t)orow * D_MODEL + ocol] = v;
                } else if (MODE == 0) {
                    ((__bf16*)outp)[(size_t)orow * D_MODEL + ocol] = (__bf16)v;
                } else {
                    const int bb = orow >> 11, s = orow & (SEQ - 1);
                    const int h  = ocol >> 6,  d = ocol & (HDIM - 1);
                    ((__bf16*)outp)[((size_t)((bb * NHEAD + h) * HDIM + d)) * SEQ + s] =
                        (__bf16)v;
                }
            }
}

// ------------------------------- attention ---------------------------------
// One block = 8 waves = 8 consecutive 16-query tiles of one (b,h).
// Per 32-key step the block stages K (32x64) and V^T (64x32) into LDS with
// GLOBAL_LOAD_ASYNC_TO_LDS_B128 (ASYNCcnt-tracked), triple-buffered so only
// one barrier per iteration is needed. Transposed flash attention:
//   S^T[k,q] = K_tile @ Q^T   ;   O^T[d,q] += V^T_tile @ P^T
__global__ __launch_bounds__(256) void attn_k(const __bf16* __restrict__ Q,
                                              const __bf16* __restrict__ K,
                                              const __bf16* __restrict__ Vt,
                                              __bf16* __restrict__ ctx) {
    __shared__ __bf16 Ktile[3][32 * 64];   // [key][d]   3 x 4 KB
    __shared__ __bf16 Vtile[3][64 * 32];   // [d][key]   3 x 4 KB
    __shared__ __bf16 Ptile[8][16 * 32];   // per-wave P^T tiles, 8 KB

    const int t     = threadIdx.x;
    const int wslot = t >> 5;
    __bf16* pl = &Ptile[wslot][0];

    const int blk  = blockIdx.x;          // 0..1023
    const int b    = blk >> 8;            // 16 heads * 16 qgroups per batch
    const int h    = (blk >> 4) & 15;
    const int qg   = blk & 15;            // 128-query group
    const int lane = t & 31;
    const int ln   = lane & 15;
    const int hf   = lane >> 4;
    const int qbase = qg * 128 + wslot * 16;

    const size_t qrow     = ((size_t)(b * SEQ + qbase + ln)) * D_MODEL + h * HDIM;
    const size_t krowbase = ((size_t)b * SEQ) * D_MODEL + h * HDIM;
    const size_t vrowbase = ((size_t)(b * NHEAD + h) * HDIM) * SEQ;

    // cooperative-staging indices (256 threads move 4KB per tile, 16B each)
    const int kkey = t >> 3, kdc = (t & 7) * 8;    // K tile
    const int vd   = t >> 2, vkc = (t & 3) * 8;    // V tile

    // issue the async copies for one 32-key tile (2 ASYNC ops per wave)
    auto issue_tile = [&](int bufi, int kt) {
        unsigned lK = (unsigned)(uintptr_t)(&Ktile[bufi][kkey * 64 + kdc]);
        unsigned lV = (unsigned)(uintptr_t)(&Vtile[bufi][vd * 32 + vkc]);
        const __bf16* gK = K + krowbase + (size_t)(kt + kkey) * D_MODEL + kdc;
        const __bf16* gV = Vt + vrowbase + (size_t)vd * SEQ + kt + vkc;
        asm volatile("global_load_async_to_lds_b128 %0, %1, off"
                     :: "v"(lK), "v"((unsigned long long)(uintptr_t)gK) : "memory");
        asm volatile("global_load_async_to_lds_b128 %0, %1, off"
                     :: "v"(lV), "v"((unsigned long long)(uintptr_t)gV) : "memory");
    };

    // Q^T B-fragments, preloaded for head-dim chunks k0=0 and k0=32
    const v16bf Bq0 = afrag(Q + qrow + 8 * hf);
    const v16bf Bq1 = afrag(Q + qrow + 32 + 8 * hf);

    v8f o[4];
#pragma unroll
    for (int dt = 0; dt < 4; ++dt)
#pragma unroll
        for (int r = 0; r < 8; ++r) o[dt][r] = 0.0f;

    float m = -1.0e30f, l = 0.0f;
    const float C = 0.125f * 1.44269504088896340736f;   // scale * log2(e)

    issue_tile(0, 0);
    int cur = 0;

#pragma unroll 1
    for (int kt = 0; kt < SEQ; kt += 32) {
        int nxt = cur + 1; if (nxt == 3) nxt = 0;
        if (kt + 32 < SEQ) {
            issue_tile(nxt, kt + 32);
            // in-order async completion: <=2 outstanding => current tile landed
            asm volatile("s_wait_asynccnt 0x2" ::: "memory");
        } else {
            asm volatile("s_wait_asynccnt 0x0" ::: "memory");
        }
        __syncthreads();   // all waves' current-tile data visible in LDS

        const __bf16* kb = &Ktile[cur][0];
        const __bf16* vb = &Vtile[cur][0];

        // ----- S^T tiles: keys kt..kt+15 (s0) and kt+16..kt+31 (s1)
        const __bf16* kpA = kb + ln * 64;
        const __bf16* kpB = kb + (16 + ln) * 64;
        v8f s0 = {}, s1 = {};
        s0 = wmma_bf16(afrag(kpA + 8 * hf),      Bq0, s0);
        s0 = wmma_bf16(afrag(kpA + 32 + 8 * hf), Bq1, s0);
        s1 = wmma_bf16(afrag(kpB + 8 * hf),      Bq0, s1);
        s1 = wmma_bf16(afrag(kpB + 32 + 8 * hf), Bq1, s1);

        // ----- online softmax over keys (register rows + one xor-16 shuffle)
        float vm = s0[0];
#pragma unroll
        for (int r = 0; r < 8; ++r) { vm = fmaxf(vm, s0[r]); vm = fmaxf(vm, s1[r]); }
        vm = fmaxf(vm, __shfl_xor(vm, 16, 32));
        const float mn    = fmaxf(m, vm);
        const float alpha = __builtin_amdgcn_exp2f((m - mn) * C);
        const float mC    = mn * C;

        float sum = 0.0f;
        v8bf p0, p1;
#pragma unroll
        for (int r = 0; r < 8; ++r) {
            float e0 = __builtin_amdgcn_exp2f(__builtin_fmaf(s0[r], C, -mC));
            float e1 = __builtin_amdgcn_exp2f(__builtin_fmaf(s1[r], C, -mC));
            sum += e0 + e1;
            p0[r] = (__bf16)e0;
            p1[r] = (__bf16)e1;
        }
        sum += __shfl_xor(sum, 16, 32);
        l = l * alpha + sum;
        m = mn;

#pragma unroll
        for (int dt = 0; dt < 4; ++dt)
#pragma unroll
            for (int r = 0; r < 8; ++r) o[dt][r] *= alpha;

        // ----- reshape P^T (C layout) -> B fragment via per-wave LDS tile
        *(v8bf*)(pl + ln * 32 + 8 * hf)      = p0;
        *(v8bf*)(pl + ln * 32 + 16 + 8 * hf) = p1;
        asm volatile("s_wait_dscnt 0" ::: "memory");
        const v16bf bp = afrag(pl + ln * 32 + 8 * hf);

        // ----- O^T accumulation: 4 d-tiles, K-dim = 32 keys
#pragma unroll
        for (int dt = 0; dt < 4; ++dt)
            o[dt] = wmma_bf16(afrag(vb + (16 * dt + ln) * 32 + 8 * hf), bp, o[dt]);

        cur = nxt;
    }

    const float inv = 1.0f / l;
    // per dt-tile, this lane's d values (16*dt + 8*hf + r) are contiguous -> b128
#pragma unroll
    for (int dt = 0; dt < 4; ++dt) {
        v8bf ov;
#pragma unroll
        for (int r = 0; r < 8; ++r) ov[r] = (__bf16)(o[dt][r] * inv);
        *(v8bf*)(ctx + ((size_t)(b * SEQ + qbase + ln)) * D_MODEL + h * HDIM +
                 16 * dt + 8 * hf) = ov;
    }
}

// ------------------------------- launcher ----------------------------------

extern "C" void kernel_launch(void* const* d_in, const int* in_sizes, int n_in,
                              void* d_out, int out_size, void* d_ws, size_t ws_size,
                              hipStream_t stream) {
    (void)in_sizes; (void)n_in; (void)out_size; (void)ws_size;

    const float* x  = (const float*)d_in[0];
    const float* Wq = (const float*)d_in[1];
    const float* bq = (const float*)d_in[2];
    const float* Wk = (const float*)d_in[3];
    const float* bk = (const float*)d_in[4];
    const float* Wv = (const float*)d_in[5];
    const float* bv = (const float*)d_in[6];
    const float* Wo = (const float*)d_in[7];
    const float* bo = (const float*)d_in[8];

    char* ws = (char*)d_ws;
    const size_t XBF  = (size_t)MROWS * D_MODEL * 2;       // 16 MiB
    const size_t WBF  = (size_t)D_MODEL * D_MODEL * 2;     // 2 MiB
    __bf16* xbf = (__bf16*)(ws);
    __bf16* Wtq = (__bf16*)(ws + XBF);
    __bf16* Wtk = (__bf16*)(ws + XBF + WBF);
    __bf16* Wtv = (__bf16*)(ws + XBF + 2 * WBF);
    __bf16* Wto = (__bf16*)(ws + XBF + 3 * WBF);
    __bf16* Qb  = (__bf16*)(ws + XBF + 4 * WBF);
    __bf16* Kb  = (__bf16*)(ws + 2 * XBF + 4 * WBF);
    __bf16* Vt  = (__bf16*)(ws + 3 * XBF + 4 * WBF);
    __bf16* ctx = (__bf16*)(ws + 4 * XBF + 4 * WBF);

    const int NX = MROWS * D_MODEL;                        // 8388608
    f32_to_bf16_k<<<NX / 256, 256, 0, stream>>>(x, xbf, NX);
    transpose_bf16_k<<<(D_MODEL * D_MODEL) / 256, 256, 0, stream>>>(Wq, Wtq);
    transpose_bf16_k<<<(D_MODEL * D_MODEL) / 256, 256, 0, stream>>>(Wk, Wtk);
    transpose_bf16_k<<<(D_MODEL * D_MODEL) / 256, 256, 0, stream>>>(Wv, Wtv);
    transpose_bf16_k<<<(D_MODEL * D_MODEL) / 256, 256, 0, stream>>>(Wo, Wto);

    // 2048 waves per GEMM (each owns a 64x64 output block) -> 256 blocks
    gemm_proj<0><<<256, 256, 0, stream>>>(xbf, Wtq, bq, (void*)Qb);
    gemm_proj<0><<<256, 256, 0, stream>>>(xbf, Wtk, bk, (void*)Kb);
    gemm_proj<1><<<256, 256, 0, stream>>>(xbf, Wtv, bv, (void*)Vt);

    attn_k<<<1024, 256, 0, stream>>>(Qb, Kb, Vt, ctx);

    gemm_proj<2><<<256, 256, 0, stream>>>(ctx, Wto, bo, d_out);
}
